// DiagGaussian_67594195304465
// MI455X (gfx1250) — matrix operand order repaired and checked by reading, not compiled, not run on hardware
//
#include <hip/hip_runtime.h>
#include <hip/hip_bf16.h>

#define BATCH   16384
#define NIN     1024
#define NOUT    512
#define NEXP    8

#define TM 32            // rows per block tile
#define TN 128           // cols per block tile
#define TK 32            // K per step (bf16 WMMA K)
#define LDA 40           // padded LDS stride in halfs (80B: 16B-aligned, bank-friendly)
#define LDB 40
#define NK (NIN / TK)           // 32 K-steps
#define MT_PER_E (BATCH / TM)   // 512 m-tiles per expert
#define NT (NOUT / TN)          // 4 n-tiles

typedef __attribute__((ext_vector_type(8)))  __bf16 v8bf;
typedef __attribute__((ext_vector_type(16))) __bf16 v16bf;
typedef __attribute__((ext_vector_type(8)))  float  v8f;
typedef __attribute__((ext_vector_type(4)))  float  v4f;

// Split two adjacent floats into packed bf16 hi-pair + lo-pair (residual).
// hi uses round-half-away (+0x8000): any hi rounding error is recaptured
// exactly in lo, so RNE tie-break logic is unnecessary.
// v_perm_b32 gathers the two high halves in a single instruction.
__device__ __forceinline__ void split_pack(float a, float b, unsigned& hi, unsigned& lo) {
    unsigned ra = __float_as_uint(a) + 0x8000u;       // bf16 in bits[31:16]
    unsigned rb = __float_as_uint(b) + 0x8000u;
    hi = __builtin_amdgcn_perm(rb, ra, 0x07060302u);  // {rb[31:16], ra[31:16]}
    float ha = __uint_as_float(ra & 0xFFFF0000u);
    float hb = __uint_as_float(rb & 0xFFFF0000u);
    unsigned la = __float_as_uint(a - ha);            // residual, truncated to bf16
    unsigned lb = __float_as_uint(b - hb);
    lo = __builtin_amdgcn_perm(lb, la, 0x07060302u);
}

// ---- kernel 1: zero per-expert counts ----
__global__ void moe_zero_counts(unsigned* counts) {
    if (threadIdx.x < NEXP) counts[threadIdx.x] = 0u;
}

// ---- kernel 2: bucket rows by expert ----
__global__ void moe_scatter(const int* __restrict__ index,
                            unsigned* __restrict__ counts,
                            int* __restrict__ bucket) {
    int i = blockIdx.x * blockDim.x + threadIdx.x;
    if (i < BATCH) {
        int e = index[i] & (NEXP - 1);
        unsigned p = atomicAdd(&counts[e], 1u);
        bucket[e * BATCH + p] = i;
    }
}

// ---- kernel 3: expert-grouped GEMM, split-bf16 WMMA, double-buffered LDS ----
__global__ __launch_bounds__(128) void moe_gemm(
    const float* __restrict__ x, const float* __restrict__ W,
    const float* __restrict__ bias, const int* __restrict__ bucket,
    const unsigned* __restrict__ counts, float* __restrict__ out_mean)
{
    const int e  = blockIdx.x / MT_PER_E;
    const int mt = blockIdx.x % MT_PER_E;
    const unsigned cnt = counts[e];
    const int rowBase = mt * TM;
    if ((unsigned)rowBase >= cnt) return;
    const int colBase = blockIdx.y * TN;

    // double-buffered LDS, stored as packed u32 (2 bf16 along k)
    __shared__ __align__(16) unsigned Ah32[2][TM * LDA / 2];
    __shared__ __align__(16) unsigned Al32[2][TM * LDA / 2];
    __shared__ __align__(16) unsigned Bh32[2][TN * LDB / 2];
    __shared__ __align__(16) unsigned Bl32[2][TN * LDB / 2];
    __shared__ int rowids[TM];

    const int tid  = threadIdx.x;
    const int lane = tid & 31;
    const int wave = tid >> 5;
    const int mi   = wave & 1;          // 16-row subtile
    const int njB  = (wave >> 1) * 4;   // first of 4 16-col subtiles
    const int lr   = lane & 15;
    const int hh   = lane >> 4;         // half-lane group

    if (tid < TM) {
        int g = rowBase + tid;
        rowids[tid] = bucket[e * BATCH + (((unsigned)g < cnt) ? g : rowBase)];
    }
    __syncthreads();

    // A-fill: thread loads 8 contiguous k of one row
    const int ar = tid >> 2;            // row 0..31
    const int ak = (tid & 3) * 8;       // k offset 0,8,16,24
    // B-fill: thread loads 4 cols x 2 adjacent k-rows, 4 iterations
    const int bc   = (tid & 31) * 4;    // col group
    const int bkw2 = (tid >> 5) * 2;    // k-row pair base (even)

    const float* xrow  = x + (size_t)rowids[ar] * NIN + ak;
    const float* wbase = W + (size_t)e * NIN * NOUT + colBase + bc;

    v8f acc0 = 0.f, acc1 = 0.f, acc2 = 0.f, acc3 = 0.f;
    v8f* accs[4] = { &acc0, &acc1, &acc2, &acc3 };

    float af[8];
    v4f bw[8];

    // ---- prologue: load + stage tile 0 into buffer 0 ----
    {
        v4f ax0 = *(const v4f*)(xrow);
        v4f ax1 = *(const v4f*)(xrow + 4);
        #pragma unroll
        for (int i = 0; i < 4; ++i) { af[i] = ax0[i]; af[4 + i] = ax1[i]; }
        #pragma unroll
        for (int it = 0; it < 4; ++it) {
            const int kr = it * 8 + bkw2;
            bw[it * 2]     = *(const v4f*)(wbase + (size_t)kr * NOUT);
            bw[it * 2 + 1] = *(const v4f*)(wbase + (size_t)(kr + 1) * NOUT);
        }
        #pragma unroll
        for (int p = 0; p < 4; ++p)
            split_pack(af[2 * p], af[2 * p + 1],
                       Ah32[0][ar * (LDA / 2) + ak / 2 + p],
                       Al32[0][ar * (LDA / 2) + ak / 2 + p]);
        #pragma unroll
        for (int it = 0; it < 4; ++it) {
            const int kr = it * 8 + bkw2;
            #pragma unroll
            for (int j = 0; j < 4; ++j)
                split_pack(bw[it * 2][j], bw[it * 2 + 1][j],
                           Bh32[0][(bc + j) * (LDB / 2) + kr / 2],
                           Bl32[0][(bc + j) * (LDB / 2) + kr / 2]);
        }
    }

    for (int kt = 0; kt < NK; ++kt) {
        __syncthreads();
        const int buf = kt & 1;
        const bool more = (kt + 1) < NK;

        // ---- issue next tile's global loads early (latency overlap) ----
        if (more) {
            const int k0 = (kt + 1) * TK;
            v4f ax0 = *(const v4f*)(xrow + k0);
            v4f ax1 = *(const v4f*)(xrow + k0 + 4);
            #pragma unroll
            for (int i = 0; i < 4; ++i) { af[i] = ax0[i]; af[4 + i] = ax1[i]; }
            #pragma unroll
            for (int it = 0; it < 4; ++it) {
                const int kr = k0 + it * 8 + bkw2;
                bw[it * 2]     = *(const v4f*)(wbase + (size_t)kr * NOUT);
                bw[it * 2 + 1] = *(const v4f*)(wbase + (size_t)(kr + 1) * NOUT);
            }
        }

        // ---- WMMA on current buffer ----
        {
            const unsigned short* AhS = (const unsigned short*)Ah32[buf];
            const unsigned short* AlS = (const unsigned short*)Al32[buf];
            const unsigned short* BhS = (const unsigned short*)Bh32[buf];
            const unsigned short* BlS = (const unsigned short*)Bl32[buf];

            const unsigned short* pAh = AhS + (mi * 16 + lr) * LDA + hh * 8;
            const unsigned short* pAl = AlS + (mi * 16 + lr) * LDA + hh * 8;
            v16bf a_h = __builtin_shufflevector(*(const v8bf*)pAh, *(const v8bf*)(pAh + 16),
                                                0,1,2,3,4,5,6,7,8,9,10,11,12,13,14,15);
            v16bf a_l = __builtin_shufflevector(*(const v8bf*)pAl, *(const v8bf*)(pAl + 16),
                                                0,1,2,3,4,5,6,7,8,9,10,11,12,13,14,15);
            #pragma unroll
            for (int j = 0; j < 4; ++j) {
                const unsigned short* pBh = BhS + ((njB + j) * 16 + lr) * LDB + hh * 8;
                const unsigned short* pBl = BlS + ((njB + j) * 16 + lr) * LDB + hh * 8;
                v16bf b_h = __builtin_shufflevector(*(const v8bf*)pBh, *(const v8bf*)(pBh + 16),
                                                    0,1,2,3,4,5,6,7,8,9,10,11,12,13,14,15);
                v16bf b_l = __builtin_shufflevector(*(const v8bf*)pBl, *(const v8bf*)(pBl + 16),
                                                    0,1,2,3,4,5,6,7,8,9,10,11,12,13,14,15);
                v8f c = *accs[j];
                c = __builtin_amdgcn_wmma_f32_16x16x32_bf16(false, a_h, false, b_h, (short)0, c, false, false);
                c = __builtin_amdgcn_wmma_f32_16x16x32_bf16(false, a_h, false, b_l, (short)0, c, false, false);
                c = __builtin_amdgcn_wmma_f32_16x16x32_bf16(false, a_l, false, b_h, (short)0, c, false, false);
                *accs[j] = c;
            }
        }

        // ---- convert + stage next tile into alternate buffer ----
        if (more) {
            const int nbuf = buf ^ 1;
            #pragma unroll
            for (int p = 0; p < 4; ++p)
                split_pack(af[2 * p], af[2 * p + 1],
                           Ah32[nbuf][ar * (LDA / 2) + ak / 2 + p],
                           Al32[nbuf][ar * (LDA / 2) + ak / 2 + p]);
            #pragma unroll
            for (int it = 0; it < 4; ++it) {
                const int kr = it * 8 + bkw2;
                #pragma unroll
                for (int j = 0; j < 4; ++j)
                    split_pack(bw[it * 2][j], bw[it * 2 + 1][j],
                               Bh32[nbuf][(bc + j) * (LDB / 2) + kr / 2],
                               Bl32[nbuf][(bc + j) * (LDB / 2) + kr / 2]);
            }
        }
    }

    // ---- epilogue: bias + scattered stores ----
    const bool full = (unsigned)(rowBase + TM) <= cnt;
    if (full) {
        #pragma unroll
        for (int j = 0; j < 4; ++j) {
            const int col = colBase + (njB + j) * 16 + lr;
            const float bv = bias[e * NOUT + col];
            const v8f c = *accs[j];
            #pragma unroll
            for (int r = 0; r < 8; ++r) {
                const int m = mi * 16 + hh * 8 + r;
                out_mean[(size_t)rowids[m] * NOUT + col] = c[r] + bv;
            }
        }
    } else {
        #pragma unroll
        for (int j = 0; j < 4; ++j) {
            const int col = colBase + (njB + j) * 16 + lr;
            const float bv = bias[e * NOUT + col];
            const v8f c = *accs[j];
            #pragma unroll
            for (int r = 0; r < 8; ++r) {
                const int m = mi * 16 + hh * 8 + r;
                if ((unsigned)(rowBase + m) < cnt)
                    out_mean[(size_t)rowids[m] * NOUT + col] = c[r] + bv;
            }
        }
    }
}

// ---- kernel 4: action_std = exp(logstd) broadcast, vectorized ----
__global__ void moe_std(const float* __restrict__ logstd, float* __restrict__ out_std) {
    size_t i4 = ((size_t)blockIdx.x * blockDim.x + threadIdx.x) * 4;
    int c = (int)(i4 & (NOUT - 1));
    v4f v;
    v.x = expf(logstd[c]);
    v.y = expf(logstd[c + 1]);
    v.z = expf(logstd[c + 2]);
    v.w = expf(logstd[c + 3]);
    *(v4f*)(out_std + i4) = v;
}

extern "C" void kernel_launch(void* const* d_in, const int* in_sizes, int n_in,
                              void* d_out, int out_size, void* d_ws, size_t ws_size,
                              hipStream_t stream) {
    const float* x      = (const float*)d_in[0];
    const int*   index  = (const int*)d_in[1];
    const float* W      = (const float*)d_in[2];
    const float* bias   = (const float*)d_in[3];
    const float* logstd = (const float*)d_in[4];
    float* out = (float*)d_out;

    unsigned* counts = (unsigned*)d_ws;
    int* bucket = (int*)d_ws + 16;   // NEXP*BATCH ints

    moe_zero_counts<<<1, 32, 0, stream>>>(counts);
    moe_scatter<<<BATCH / 256, 256, 0, stream>>>(index, counts, bucket);
    moe_gemm<<<dim3(NEXP * MT_PER_E, NT), 128, 0, stream>>>(x, W, bias, bucket, counts, out);
    moe_std<<<(BATCH * NOUT) / 4 / 256, 256, 0, stream>>>(logstd, out + (size_t)BATCH * NOUT);
}
